// Net_Undpp_81690277970269
// MI455X (gfx1250) — compile-verified, matrix operand order checked by reading, not run on hardware
//
#include <hip/hip_runtime.h>

// ---------------------------------------------------------------------------
// MI455X (gfx1250, wave32). WMMA (v_wmma_f32_16x16x32_f16, f32 accum) for the
// GEMM-shaped stages, 2 M-tiles per wave (B-fragment + K-index math reuse);
// branchless clamped gathers for the fused disp_shift A-operand; weight panels
// staged to LDS in fragment order via global_load_async_to_lds (ASYNCcnt);
// activations packed NHWC-f16 so the pointwise A-operand is aligned b128s.
// ---------------------------------------------------------------------------

typedef __attribute__((ext_vector_type(16))) _Float16 v16h;
typedef __attribute__((ext_vector_type(8)))  _Float16 v8h;
typedef __attribute__((ext_vector_type(8)))  float    v8f;

#define ANGR 9
#define HS   96
#define WS   96
#define HU   864
#define WV   864
#define BN2  2
#define C0   7
#define CC   144
#define CDCH 180
#define SPX  (HS*WS)          /* 9216  */
#define SUX  (HU*WV)          /* 746496 */
#define KBC  567              /* 7*9*9 */
#define EPSB 1e-5f

#if defined(__gfx1250__) && __has_builtin(__builtin_amdgcn_global_load_async_to_lds_b32) && __has_builtin(__builtin_amdgcn_s_wait_asynccnt)
#define USE_ASYNC_LDS 1
typedef int __attribute__((address_space(1)))* gas_ip;   // global b32 operand
typedef int __attribute__((address_space(3)))* las_ip;   // LDS b32 operand
#else
#define USE_ASYNC_LDS 0
#endif

// ---------------- SAI -> MacPI rearrange ----------------
__global__ __launch_bounds__(256) void k_sai2macpi(const float* __restrict__ x,
                                                   float* __restrict__ o) {
  int idx = blockIdx.x * blockDim.x + threadIdx.x;
  if (idx >= BN2 * SUX) return;
  int b = idx / SUX, r = idx % SUX;
  int p = r / WV, q = r % WV;
  int i = p / ANGR, u = p % ANGR, j = q / ANGR, v = q % ANGR;
  o[idx] = x[b * SUX + (u * HS + i) * WV + (v * WS + j)];
}

// ---------------- LDS-tiled 3x3 conv, dilation=pad=9, Cout=7 ----------------
__global__ __launch_bounds__(256) void k_conv3x3_d9(const float* __restrict__ in,
                                                    const float* __restrict__ w,
                                                    float* __restrict__ out,
                                                    int Cin) {
  __shared__ float tile[34 * 35];
  __shared__ float wsh[C0 * C0 * 9];
  const int tid = threadIdx.x;
  const int b = blockIdx.y;
  const int ti = blockIdx.x / 54, tj = blockIdx.x % 54;
  const int ty0 = ti * 16, tx0 = tj * 16;
  const int ly = tid >> 4, lx = tid & 15;
  const int p = ty0 + ly, q = tx0 + lx;

  for (int t = tid; t < C0 * Cin * 9; t += 256) wsh[t] = w[t];

  float acc[C0];
#pragma unroll
  for (int oc = 0; oc < C0; ++oc) acc[oc] = 0.f;

  for (int ic = 0; ic < Cin; ++ic) {
    __syncthreads();
    const float* ip = in + ((long)b * Cin + ic) * SUX;
    for (int t = tid; t < 34 * 34; t += 256) {
      int r = t / 34, c = t - r * 34;
      int gr = ty0 - 9 + r, gc = tx0 - 9 + c;
      float v = 0.f;
      if (gr >= 0 && gr < HU && gc >= 0 && gc < WV) v = ip[(long)gr * WV + gc];
      tile[r * 35 + c] = v;
    }
    __syncthreads();
    float tv[9];
#pragma unroll
    for (int ku = 0; ku < 3; ++ku)
#pragma unroll
      for (int kv = 0; kv < 3; ++kv)
        tv[ku * 3 + kv] = tile[(ly + 9 * ku) * 35 + (lx + 9 * kv)];
#pragma unroll
    for (int oc = 0; oc < C0; ++oc) {
      const float* wp = &wsh[(oc * Cin + ic) * 9];
#pragma unroll
      for (int t9 = 0; t9 < 9; ++t9) acc[oc] = fmaf(tv[t9], wp[t9], acc[oc]);
    }
  }
#pragma unroll
  for (int oc = 0; oc < C0; ++oc)
    out[((long)b * C0 + oc) * SUX + (long)p * WV + q] = acc[oc];
}

// ---------------- training-mode BN stats (one block per channel) -------------
__global__ __launch_bounds__(256) void k_bnstats(const float* __restrict__ x,
                                                 float* __restrict__ stats,
                                                 int C, long S) {
  int c = blockIdx.x;
  float s = 0.f, s2 = 0.f;
  for (int b = 0; b < BN2; ++b) {
    const float* xp = x + ((long)b * C + c) * S;
    for (long i = threadIdx.x; i < S; i += blockDim.x) {
      float v = xp[i];
      s += v; s2 += v * v;
    }
  }
  __shared__ float sh[256], sh2[256];
  sh[threadIdx.x] = s; sh2[threadIdx.x] = s2;
  __syncthreads();
  for (int st = 128; st > 0; st >>= 1) {
    if ((int)threadIdx.x < st) {
      sh[threadIdx.x]  += sh[threadIdx.x + st];
      sh2[threadIdx.x] += sh2[threadIdx.x + st];
    }
    __syncthreads();
  }
  if (threadIdx.x == 0) {
    float n = (float)(BN2 * S);
    float m = sh[0] / n;
    float var = sh2[0] / n - m * m;          // biased batch variance
    stats[2 * c]     = m;
    stats[2 * c + 1] = rsqrtf(var + EPSB);
  }
}

// ---------------- fused BN normalize + ReLU (in place) ----------------
__global__ __launch_bounds__(256) void k_bnrelu(float* __restrict__ x,
                                                const float* __restrict__ stats,
                                                const float* __restrict__ g,
                                                const float* __restrict__ bta,
                                                int C, long S) {
  long idx = (long)blockIdx.x * blockDim.x + threadIdx.x;
  if (idx >= (long)BN2 * C * S) return;
  int c = (int)((idx / S) % C);
  float v = (x[idx] - stats[2 * c]) * stats[2 * c + 1] * g[c] + bta[c];
  x[idx] = fmaxf(v, 0.f);
}

// ---------------- pack NCHW f32 -> NHWC f16 (zero-padded channels) -----------
__global__ __launch_bounds__(256) void k_packT(const float* __restrict__ x,
                                               _Float16* __restrict__ xt,
                                               int C, int Cpad) {
  long idx = (long)blockIdx.x * blockDim.x + threadIdx.x;
  if (idx >= (long)BN2 * SPX * Cpad) return;
  int c = (int)(idx % Cpad); long t = idx / Cpad;
  int s = (int)(t % SPX);
  int b = (int)(t / SPX);
  float v = (c < C) ? x[((long)b * C + c) * SPX + s] : 0.f;
  xt[idx] = (_Float16)v;
}

// ---------------- BuildCost: fused disp_shift + grouped 9x9/s9 conv ----------
// Per disparity d: GEMM M=18432 (pixels), N=16 (group out ch), K=567 (7*81).
// 2 M-tiles per wave; branchless clamped gathers; LDS weight panel.
__global__ __launch_bounds__(128) void k_buildcost_wmma(const float* __restrict__ f,
                                                        const float* __restrict__ bw,
                                                        float* __restrict__ cv) {
  __shared__ float Bs[18 * 512];              // [ks][lane][e] fragment order
  const int tid = threadIdx.x;
  const int wave = tid >> 5;
  const int lane = tid & 31;
  const int wt = (blockIdx.x * 4 + wave) * 2; // first of two M-tiles (0..1150)
  const int d = blockIdx.y;                   // 0..8 (dd = d-4)
  const int dd = d - 4;
  int dil, pad;
  if (dd < 0)      { dil = -dd * ANGR + 1; pad = 36 * (-dd); }
  else if (dd == 0){ dil = 1;              pad = 0; }
  else             { dil = dd * ANGR - 1;  pad = 36 * dd - 8; }

  // stage 16 x 576 weight panel (group d), fragment-ordered
  for (int t = tid; t < 18 * 512; t += 128) {
    int e = t & 15, ln = (t >> 4) & 31, ks = t >> 9;
    int k = ks * 32 + ((ln >> 4) << 3) + (e & 7) + ((e >> 3) << 4);
    int oc = d * 16 + (ln & 15);
#if USE_ASYNC_LDS
    if (k < KBC)
      __builtin_amdgcn_global_load_async_to_lds_b32(
          (gas_ip)&bw[(long)oc * KBC + k], (las_ip)&Bs[t], 0, 0);
    else
      Bs[t] = 0.f;
#else
    Bs[t] = (k < KBC) ? bw[(long)oc * KBC + k] : 0.f;
#endif
  }
#if USE_ASYNC_LDS
  __builtin_amdgcn_s_wait_asynccnt(0);
#endif
  __syncthreads();

  int i9t[2], j9t[2];
  const float* fbt[2];
#pragma unroll
  for (int t2 = 0; t2 < 2; ++t2) {
    int mrow = (wt + t2) * 16 + (lane & 15);
    int b = mrow / SPX, rem = mrow % SPX;
    i9t[t2] = (rem / WS) * ANGR;
    j9t[t2] = (rem % WS) * ANGR;
    fbt[t2] = f + (long)b * C0 * SUX;
  }
  const int kbase = (lane >> 4) << 3;

  v8f acc0 = {}, acc1 = {};
  for (int ks = 0; ks < 18; ++ks) {
    // B fragment: contiguous ds reads, convert f32->f16 (shared by both tiles)
    const float* bp = &Bs[ks * 512 + lane * 16];
    v16h A0, A1, Bv;
#pragma unroll
    for (int e = 0; e < 16; ++e) Bv[e] = (_Float16)bp[e];

    // K-index math shared across both tiles: two runs of 8 consecutive k
    const int s0 = ks * 32 + kbase;
#pragma unroll
    for (int run = 0; run < 2; ++run) {
      int s = s0 + (run << 4);
      int ic = s / 81; int r81 = s - ic * 81;
      int u = r81 / 9; int v = r81 - u * 9;
#pragma unroll
      for (int e = 0; e < 8; ++e) {
        int vv = v + e;
        int uu = u + (vv >= 9 ? 1 : 0); vv -= (vv >= 9 ? 9 : 0);
        int icc = ic + (uu >= 9 ? 1 : 0); uu -= (uu >= 9 ? 9 : 0);
        int uoff = uu * dil - pad;           // pixel-independent
        int voff = vv * dil - pad;
        bool okic = (icc < C0);
        long coff = (long)(okic ? icc : 0) * SUX;
#pragma unroll
        for (int t2 = 0; t2 < 2; ++t2) {
          int row = i9t[t2] + uoff;
          int col = j9t[t2] + voff;
          bool ok = okic & ((unsigned)row < (unsigned)HU) &
                           ((unsigned)col < (unsigned)WV);
          int rc = min(max(row, 0), HU - 1);
          int cc = min(max(col, 0), WV - 1);
          float av = fbt[t2][coff + (long)rc * WV + cc];  // unconditional (L2 hit)
          _Float16 h = ok ? (_Float16)av : (_Float16)0.f;
          if (t2 == 0) A0[(run << 3) + e] = h;
          else         A1[(run << 3) + e] = h;
        }
      }
    }
    acc0 = __builtin_amdgcn_wmma_f32_16x16x32_f16(false, A0, false, Bv,
                                                  (short)0, acc0, false, false);
    acc1 = __builtin_amdgcn_wmma_f32_16x16x32_f16(false, A1, false, Bv,
                                                  (short)0, acc1, false, false);
  }
  const int oc = d * 16 + (lane & 15);
#pragma unroll
  for (int t2 = 0; t2 < 2; ++t2) {
#pragma unroll
    for (int r = 0; r < 8; ++r) {            // D: row r+8*(lane>=16), col lane&15
      int m = (wt + t2) * 16 + r + ((lane >> 4) << 3);
      int bb = m / SPX, rm = m % SPX;
      float vv = (t2 == 0) ? acc0[r] : acc1[r];
      cv[((long)bb * CC + oc) * SPX + rm] = vv;
    }
  }
}

// ---------------- depthwise 3x3 conv, pad 1 ----------------
__global__ __launch_bounds__(256) void k_dwconv3(const float* __restrict__ in,
                                                 const float* __restrict__ w,
                                                 float* __restrict__ out, int C) {
  long idx = (long)blockIdx.x * blockDim.x + threadIdx.x;
  if (idx >= (long)BN2 * C * SPX) return;
  int j = (int)(idx % WS); long t = idx / WS;
  int i = (int)(t % HS);   t /= HS;
  int c = (int)(t % C);
  int b = (int)(t / C);
  const float* ip = in + ((long)b * C + c) * SPX;
  const float* wp = w + (long)c * 9;
  float acc = 0.f;
#pragma unroll
  for (int ku = 0; ku < 3; ++ku) {
    int r = i + ku - 1;
    if (r < 0 || r >= HS) continue;
#pragma unroll
    for (int kv = 0; kv < 3; ++kv) {
      int cc = j + kv - 1;
      if (cc < 0 || cc >= WS) continue;
      acc = fmaf(ip[r * WS + cc], wp[ku * 3 + kv], acc);
    }
  }
  out[idx] = acc;
}

// ---------------- pointwise 1x1 conv via WMMA, 2 M-tiles per wave ------------
__global__ __launch_bounds__(128) void k_pw_wmma(const _Float16* __restrict__ xt,
                                                 const float* __restrict__ w,
                                                 float* __restrict__ out,
                                                 int Cin, int Cout) {
  __shared__ float Bs[6 * 512];
  const int tid = threadIdx.x;
  const int wave = tid >> 5;
  const int lane = tid & 31;
  const int wt = (blockIdx.x * 4 + wave) * 2; // first of two M-tiles
  const int nt = blockIdx.y;
  const int ksteps = (Cin + 31) >> 5;
  const int Cpad = ksteps << 5;

  for (int t = tid; t < ksteps * 512; t += 128) {
    int e = t & 15, ln = (t >> 4) & 31, ks = t >> 9;
    int k = ks * 32 + ((ln >> 4) << 3) + (e & 7) + ((e >> 3) << 4);
    int oc = nt * 16 + (ln & 15);
#if USE_ASYNC_LDS
    if (k < Cin && oc < Cout)
      __builtin_amdgcn_global_load_async_to_lds_b32(
          (gas_ip)&w[(long)oc * Cin + k], (las_ip)&Bs[t], 0, 0);
    else
      Bs[t] = 0.f;
#else
    Bs[t] = (k < Cin && oc < Cout) ? w[(long)oc * Cin + k] : 0.f;
#endif
  }
#if USE_ASYNC_LDS
  __builtin_amdgcn_s_wait_asynccnt(0);
#endif
  __syncthreads();

  const int kbase = (lane >> 4) << 3;
  const _Float16* ab0 = xt + (long)(wt * 16 + (lane & 15)) * Cpad;
  const _Float16* ab1 = ab0 + (long)16 * Cpad;   // next tile, same lane row

  v8f acc0 = {}, acc1 = {};
  for (int ks = 0; ks < ksteps; ++ks) {
    const float* bp = &Bs[ks * 512 + lane * 16];
    v16h A0, A1, Bv;
#pragma unroll
    for (int e = 0; e < 16; ++e) Bv[e] = (_Float16)bp[e];
    v8h a00 = *(const v8h*)(ab0 + ks * 32 + kbase);
    v8h a01 = *(const v8h*)(ab0 + ks * 32 + kbase + 16);
    v8h a10 = *(const v8h*)(ab1 + ks * 32 + kbase);
    v8h a11 = *(const v8h*)(ab1 + ks * 32 + kbase + 16);
#pragma unroll
    for (int e = 0; e < 8; ++e) {
      A0[e] = a00[e]; A0[e + 8] = a01[e];
      A1[e] = a10[e]; A1[e + 8] = a11[e];
    }
    acc0 = __builtin_amdgcn_wmma_f32_16x16x32_f16(false, A0, false, Bv,
                                                  (short)0, acc0, false, false);
    acc1 = __builtin_amdgcn_wmma_f32_16x16x32_f16(false, A1, false, Bv,
                                                  (short)0, acc1, false, false);
  }
  const int oc = nt * 16 + (lane & 15);
  if (oc < Cout) {
#pragma unroll
    for (int t2 = 0; t2 < 2; ++t2) {
#pragma unroll
      for (int r = 0; r < 8; ++r) {
        int m = (wt + t2) * 16 + r + ((lane >> 4) << 3);
        int bb = m / SPX, rm = m % SPX;
        float vv = (t2 == 0) ? acc0[r] : acc1[r];
        out[((long)bb * Cout + oc) * SPX + rm] = vv;
      }
    }
  }
}

// ---------------- last 1x1 (180->9) + softmax + disparity expectation --------
__global__ __launch_bounds__(256) void k_final(const float* __restrict__ y,
                                               const float* __restrict__ lw,
                                               float* __restrict__ out) {
  int idx = blockIdx.x * blockDim.x + threadIdx.x;
  if (idx >= BN2 * SPX) return;
  int b = idx / SPX, rm = idx % SPX;
  const float* yb = y + (long)b * CDCH * SPX + rm;
  float c[9];
#pragma unroll
  for (int ch = 0; ch < 9; ++ch) c[ch] = 0.f;
  for (int k = 0; k < CDCH; ++k) {
    float v = yb[(long)k * SPX];
#pragma unroll
    for (int ch = 0; ch < 9; ++ch) c[ch] = fmaf(v, lw[ch * CDCH + k], c[ch]);
  }
  float mx = c[0];
#pragma unroll
  for (int ch = 1; ch < 9; ++ch) mx = fmaxf(mx, c[ch]);
  float s = 0.f, ex = 0.f;
#pragma unroll
  for (int ch = 0; ch < 9; ++ch) {
    float e = __expf(c[ch] - mx);
    s += e; ex += e * (float)(ch - 4);
  }
  out[idx] = ex / s;
}

// ---------------------------------------------------------------------------
extern "C" void kernel_launch(void* const* d_in, const int* in_sizes, int n_in,
                              void* d_out, int out_size, void* d_ws, size_t ws_size,
                              hipStream_t stream) {
  const float* x       = (const float*)d_in[0];
  const float* fw0     = (const float*)d_in[1];
  const float* fg0     = (const float*)d_in[2];
  const float* fb0     = (const float*)d_in[3];
  const float* fw      = (const float*)d_in[4];
  const float* fg      = (const float*)d_in[5];
  const float* fbv     = (const float*)d_in[6];
  const float* bc_w    = (const float*)d_in[7];
  const float* agg0_dw = (const float*)d_in[8];
  const float* agg0_dg = (const float*)d_in[9];
  const float* agg0_db = (const float*)d_in[10];
  const float* agg0_pw = (const float*)d_in[11];
  const float* agg0_pg = (const float*)d_in[12];
  const float* agg0_pb = (const float*)d_in[13];
  const float* aggm_dw = (const float*)d_in[14];
  const float* aggm_dg = (const float*)d_in[15];
  const float* aggm_db = (const float*)d_in[16];
  const float* aggm_pw = (const float*)d_in[17];
  const float* aggm_pg = (const float*)d_in[18];
  const float* aggm_pb = (const float*)d_in[19];
  const float* last_dw = (const float*)d_in[20];
  const float* last_dg = (const float*)d_in[21];
  const float* last_db = (const float*)d_in[22];
  const float* last_pw = (const float*)d_in[23];
  float* outp = (float*)d_out;

  // workspace layout (bytes)
  const size_t MACPI_B = (size_t)BN2 * SUX * 4;           //  5,971,968
  const size_t FEAT_B  = (size_t)BN2 * C0 * SUX * 4;      // 41,803,776
  const size_t CV_B    = (size_t)BN2 * CC * SPX * 4;      // 10,616,832
  const size_t Y_B     = (size_t)BN2 * CDCH * SPX * 4;    // 13,271,040
  char* ws = (char*)d_ws;
  float* macpi = (float*)(ws);
  float* fA    = (float*)(ws + MACPI_B);
  float* fB    = (float*)(ws + MACPI_B + FEAT_B);
  float* stats = (float*)(ws + MACPI_B + 2 * FEAT_B);
  _Float16* xt = (_Float16*)(ws + MACPI_B + 2 * FEAT_B + 4096); // <= 7,077,888 B
  // cv/y0/y1 alias fB (free after last feature conv reads it)
  float* cv = fB;
  float* y0 = (float*)((char*)fB + CV_B);
  float* y1 = (float*)((char*)y0 + Y_B);

  const long NFEAT = (long)BN2 * C0 * SUX;   // 10,450,944
  const long NY    = (long)BN2 * CDCH * SPX; // 3,317,760
  const long NCV   = (long)BN2 * CC * SPX;   // 2,654,208
  const dim3 gconv(54 * 54, BN2);

  // 1) SAI -> MacPI
  k_sai2macpi<<<(BN2 * SUX + 255) / 256, 256, 0, stream>>>(x, macpi);

  // 2) feature extraction: tiled conv(dil=pad=9) + BN(batch stats) + ReLU, x7
  k_conv3x3_d9<<<gconv, 256, 0, stream>>>(macpi, fw0, fA, 1);
  k_bnstats<<<C0, 256, 0, stream>>>(fA, stats, C0, SUX);
  k_bnrelu<<<(int)((NFEAT + 255) / 256), 256, 0, stream>>>(fA, stats, fg0, fb0, C0, SUX);

  float* src = fA; float* dst = fB;
  for (int l = 0; l < 6; ++l) {
    k_conv3x3_d9<<<gconv, 256, 0, stream>>>(src, fw + (long)l * C0 * C0 * 9, dst, C0);
    k_bnstats<<<C0, 256, 0, stream>>>(dst, stats, C0, SUX);
    k_bnrelu<<<(int)((NFEAT + 255) / 256), 256, 0, stream>>>(
        dst, stats, fg + l * C0, fbv + l * C0, C0, SUX);
    float* t = src; src = dst; dst = t;
  }
  // final features in `src` (== fA); fB becomes scratch.

  // 3) BuildCost (fused disp_shift): WMMA GEMM per disparity group
  {
    dim3 g(144, 9);   // 144 blocks * 4 waves * 2 tiles = 1152 M-tiles
    k_buildcost_wmma<<<g, 128, 0, stream>>>(src, bc_w, cv);
  }

  // 4) aggregation head
  k_dwconv3<<<(int)((NCV + 255) / 256), 256, 0, stream>>>(cv, agg0_dw, y0, CC);
  k_bnstats<<<CC, 256, 0, stream>>>(y0, stats, CC, SPX);
  k_bnrelu<<<(int)((NCV + 255) / 256), 256, 0, stream>>>(y0, stats, agg0_dg, agg0_db, CC, SPX);
  {
    int Cpad = (((CC + 31) >> 5) << 5);      // 160
    long n = (long)BN2 * SPX * Cpad;
    k_packT<<<(int)((n + 255) / 256), 256, 0, stream>>>(y0, xt, CC, Cpad);
    dim3 g(144, (CDCH + 15) / 16);
    k_pw_wmma<<<g, 128, 0, stream>>>(xt, agg0_pw, y1, CC, CDCH);
  }
  k_bnstats<<<CDCH, 256, 0, stream>>>(y1, stats, CDCH, SPX);
  k_bnrelu<<<(int)((NY + 255) / 256), 256, 0, stream>>>(y1, stats, agg0_pg, agg0_pb, CDCH, SPX);

  for (int l = 0; l < 4; ++l) {
    k_dwconv3<<<(int)((NY + 255) / 256), 256, 0, stream>>>(
        y1, aggm_dw + (long)l * CDCH * 9, y0, CDCH);
    k_bnstats<<<CDCH, 256, 0, stream>>>(y0, stats, CDCH, SPX);
    k_bnrelu<<<(int)((NY + 255) / 256), 256, 0, stream>>>(
        y0, stats, aggm_dg + l * CDCH, aggm_db + l * CDCH, CDCH, SPX);
    int Cpad = (((CDCH + 31) >> 5) << 5);    // 192
    long n = (long)BN2 * SPX * Cpad;
    k_packT<<<(int)((n + 255) / 256), 256, 0, stream>>>(y0, xt, CDCH, Cpad);
    dim3 g(144, (CDCH + 15) / 16);
    k_pw_wmma<<<g, 128, 0, stream>>>(xt, aggm_pw + (long)l * CDCH * CDCH, y1, CDCH, CDCH);
    k_bnstats<<<CDCH, 256, 0, stream>>>(y1, stats, CDCH, SPX);
    k_bnrelu<<<(int)((NY + 255) / 256), 256, 0, stream>>>(
        y1, stats, aggm_pg + l * CDCH, aggm_pb + l * CDCH, CDCH, SPX);
  }

  k_dwconv3<<<(int)((NY + 255) / 256), 256, 0, stream>>>(y1, last_dw, y0, CDCH);
  k_bnstats<<<CDCH, 256, 0, stream>>>(y0, stats, CDCH, SPX);
  k_bnrelu<<<(int)((NY + 255) / 256), 256, 0, stream>>>(y0, stats, last_dg, last_db, CDCH, SPX);

  // 5) last pw (180->9) + softmax + disparity expectation, fused
  k_final<<<(BN2 * SPX + 255) / 256, 256, 0, stream>>>(y0, last_pw, outp);
}
